// Attention_55637006352558
// MI455X (gfx1250) — matrix operand order repaired and checked by reading, not compiled, not run on hardware
//
#include <hip/hip_runtime.h>
#include <hip/hip_bf16.h>

// ---------------------------------------------------------------------------
// Attention block with fused LoRA for MI455X (gfx1250, wave32, WMMA).
// GEMMs: block-cooperative 256x64 tiles, weight tiles staged into LDS via the
// Tensor Data Mover (qkv) / async global->LDS loads (proj), double-buffered.
// ---------------------------------------------------------------------------

typedef __attribute__((ext_vector_type(16))) _Float16 v16h;
typedef __attribute__((ext_vector_type(8)))  _Float16 v8h;
typedef __attribute__((ext_vector_type(8)))  float    v8f;
typedef __attribute__((ext_vector_type(4)))  int      v4i;
typedef unsigned int uint32x4 __attribute__((ext_vector_type(4)));
typedef int          int32x8  __attribute__((ext_vector_type(8)));
typedef int          int32x4  __attribute__((ext_vector_type(4)));

typedef __attribute__((address_space(1))) v4i* gptr_b128;   // global int4*
typedef __attribute__((address_space(3))) v4i* lptr_b128;   // LDS int4*

#define DIM        1024
#define QKV_DIM    3072
#define NTOK       4096     // B*N = 2*2048
#define SEQ        2048
#define NHEADS     16
#define HDIM       64
#define ATT_SCALE  0.125f   // 64^-0.5
#define LORA_R     8
#define LORA_SCALING 2.0f   // 16 / 8

#define HAS_TDM        __has_builtin(__builtin_amdgcn_tensor_load_to_lds)
#define HAS_ASYNC      __has_builtin(__builtin_amdgcn_global_load_async_to_lds_b128)
#define HAS_WAIT_ASYNC __has_builtin(__builtin_amdgcn_s_wait_asynccnt)

union V16H { v16h v; v8h h[2]; _Float16 e[16]; };

__device__ __forceinline__ v8f wmma_f16(v16h a, v16h b, v8f c) {
  return __builtin_amdgcn_wmma_f32_16x16x32_f16(
      /*neg_a=*/false, a, /*neg_b=*/false, b,
      /*c_mod=*/(short)0, c, /*reuse_a=*/false, /*reuse_b=*/false);
}

// A fragment, 16x32 f16 (MxK), row-major src at tile[0][0].
// lane l: row m=l&15, kh=l>>4; elems 0..7 -> K=kh*8+0..7, elems 8..15 -> +16.
__device__ __forceinline__ v16h load_a_frag(const _Float16* src, int ldk) {
  const int lane = threadIdx.x & 31;
  const int m = lane & 15, kh = lane >> 4;
  const _Float16* p = src + (size_t)m * ldk + kh * 8;
  V16H r;
  r.h[0] = *(const v8h*)(p);
  r.h[1] = *(const v8h*)(p + 16);
  return r.v;
}

// B fragment, 32x16 f16 (KxN) for C = X @ W^T from row-major W[n][k];
// lane l holds column n=l&15, kh=l>>4; elems 0..7 -> K=kh*8+0..7, elems 8..15 -> +16.
__device__ __forceinline__ v16h load_bT_frag(const _Float16* w, int ldk) {
  const int lane = threadIdx.x & 31;
  const int n = lane & 15, kh = lane >> 4;
  const _Float16* p = w + (size_t)n * ldk + kh * 8;
  V16H r;
  r.h[0] = *(const v8h*)(p);
  r.h[1] = *(const v8h*)(p + 16);
  return r.v;
}

// 32-bit LDS byte offset of a pointer into a __shared__ array.
__device__ __forceinline__ unsigned lds_addr32(const void* p) {
  return (unsigned)(unsigned long long)
      (const __attribute__((address_space(3))) void*)p;
}

// ---------------------------------------------------------------------------
// Stage a 64-row x 32-col f16 weight tile  W[n0..n0+63][kk..kk+31] -> LDS.
// USE_TDM=1: one TDM descriptor issued by wave 0.  Otherwise: async (or
// plain) per-thread copies, 16B per thread, 256 threads = 4KB tile.
// ---------------------------------------------------------------------------
template <int TROWS, int USE_TDM>
__device__ __forceinline__ void stage_w_tile(const _Float16* w, int n0, int kk,
                                             _Float16* dst) {
#if HAS_TDM
  if (USE_TDM) {
    if ((threadIdx.x >> 5) == 0) {
      const unsigned long long ga =
          (unsigned long long)(const void*)(w + (size_t)n0 * DIM + kk);
      uint32x4 g0;
      g0[0] = 1u;                                  // count=1, user mode
      g0[1] = lds_addr32(dst);                     // lds_addr
      g0[2] = (unsigned)(ga & 0xffffffffull);      // global_addr[31:0]
      g0[3] = (unsigned)((ga >> 32) & 0x01ffffffull) | (2u << 30); // [56:32]|type=2
      int32x8 g1;
      g1[0] = (int)(1u << 16);          // data_size=1 (2 bytes)
      g1[1] = (int)((unsigned)DIM << 16);   // tensor_dim0[15:0] @bits 63:48
      g1[2] = (int)((unsigned)TROWS << 16); // dim0 hi=0 | tensor_dim1[15:0]
      g1[3] = (int)(32u << 16);             // dim1 hi=0 | tile_dim0=32
      g1[4] = 64;                           // tile_dim1=64, tile_dim2=0
      g1[5] = DIM;                          // tensor_dim0_stride[31:0]
      g1[6] = 0;                            // stride0 hi | stride1 lo
      g1[7] = 0;
      int32x4 gz = {0, 0, 0, 0};
#if __clang_major__ >= 23
      int32x8 z8 = {0, 0, 0, 0, 0, 0, 0, 0};
      __builtin_amdgcn_tensor_load_to_lds(g0, g1, gz, gz, z8, 0);
#else
      __builtin_amdgcn_tensor_load_to_lds(g0, g1, gz, gz, 0);
#endif
    }
    return;
  }
#endif
  const int t = threadIdx.x;                        // 0..255
  const _Float16* src = w + (size_t)(n0 + (t >> 2)) * DIM + kk + (t & 3) * 8;
#if HAS_ASYNC
  __builtin_amdgcn_global_load_async_to_lds_b128(
      (gptr_b128)(void*)src,
      (lptr_b128)(void*)(dst + t * 8), 0, 0);
#else
  *(v8h*)(dst + t * 8) = *(const v8h*)src;
#endif
}

template <int USE_TDM>
__device__ __forceinline__ void stage_wait(bool has_next) {
#if HAS_TDM
  if (USE_TDM) {
    if ((threadIdx.x >> 5) == 0) {
      if (has_next) __builtin_amdgcn_s_wait_tensorcnt(1);
      else          __builtin_amdgcn_s_wait_tensorcnt(0);
    }
    return;
  }
#endif
#if HAS_ASYNC
#if HAS_WAIT_ASYNC
  if (has_next) __builtin_amdgcn_s_wait_asynccnt(1);
  else          __builtin_amdgcn_s_wait_asynccnt(0);
#else
  if (has_next) asm volatile("s_wait_asynccnt 1" ::: "memory");
  else          asm volatile("s_wait_asynccnt 0" ::: "memory");
#endif
#endif
}

// ---------------------------------------------------------------------------
// fp32 -> f16 elementwise convert
// ---------------------------------------------------------------------------
__global__ __launch_bounds__(256) void cvt_f32_to_f16(const float* __restrict__ in,
                                                      _Float16* __restrict__ out, int n) {
  int i = blockIdx.x * blockDim.x + threadIdx.x;
  if (i < n) out[i] = (_Float16)in[i];
}

// ---------------------------------------------------------------------------
// LoRA down-projection: xa[tok][r] = sum_c x_h[tok][c] * l1[r][c]   (rank 8)
// ---------------------------------------------------------------------------
__global__ __launch_bounds__(256) void lora_down(const _Float16* __restrict__ xh,
                                                 const float* __restrict__ l1,
                                                 float* __restrict__ xa) {
  int idx = blockIdx.x * blockDim.x + threadIdx.x;   // tok*8 + r
  if (idx >= NTOK * LORA_R) return;
  int tok = idx >> 3, r = idx & 7;
  const _Float16* xp = xh + (size_t)tok * DIM;
  const float*    lp = l1 + (size_t)r * DIM;
  float s = 0.f;
  #pragma unroll 8
  for (int c = 0; c < DIM; ++c) s += (float)xp[c] * lp[c];
  xa[idx] = s;
}

// ---------------------------------------------------------------------------
// QKV GEMM: qkv = x_h @ qkv_w^T + LORA_SCALING * xa1 @ l2^T
// Block: 256 thr = 8 waves; block tile 256m x 64n; wave tile 32m x 64n.
// Weight tile staged via TDM, double buffered.  Epilogue scatters into
// q [B,H,N,D] (pre-scaled), k [B,H,N,D], vT [B,H,D,N]  (f16).
// ---------------------------------------------------------------------------
__global__ __launch_bounds__(256) void qkv_gemm(const _Float16* __restrict__ xh,
                                                const _Float16* __restrict__ wh,
                                                const float* __restrict__ xa1,
                                                const float* __restrict__ l2, // [3072][8]
                                                _Float16* __restrict__ q,
                                                _Float16* __restrict__ k,
                                                _Float16* __restrict__ vT) {
  const int NBLK = QKV_DIM / 64;                  // 48
  const int mb = blockIdx.x / NBLK;
  const int nb = blockIdx.x % NBLK;
  const int wv = threadIdx.x >> 5;
  const int m0 = mb * 256 + wv * 32;
  const int n0 = nb * 64;

  __shared__ _Float16 bbuf[2][64 * 32];

  stage_w_tile<QKV_DIM, 1>(wh, n0, 0, bbuf[0]);
  v8f acc[2][4] = {};
  for (int kk = 0; kk < DIM; kk += 32) {
    const int cur = (kk >> 5) & 1;
    const bool has_next = (kk + 32) < DIM;
    if (has_next) stage_w_tile<QKV_DIM, 1>(wh, n0, kk + 32, bbuf[cur ^ 1]);
    stage_wait<1>(has_next);
    __syncthreads();
    const _Float16* ap = xh + (size_t)m0 * DIM + kk;
    __builtin_prefetch(ap + 32, 0, 1);            // global_prefetch_b8
    v16h a0 = load_a_frag(ap, DIM);
    v16h a1 = load_a_frag(ap + (size_t)16 * DIM, DIM);
    #pragma unroll
    for (int c = 0; c < 4; ++c) {
      v16h b = load_bT_frag(bbuf[cur] + c * 16 * 32, 32);
      acc[0][c] = wmma_f16(a0, b, acc[0][c]);
      acc[1][c] = wmma_f16(a1, b, acc[1][c]);
    }
    __syncthreads();
  }

  const int lane = threadIdx.x & 31;
  const int nloc = lane & 15, mh = lane >> 4;
  #pragma unroll
  for (int c = 0; c < 4; ++c) {
    const int col = n0 + c * 16 + nloc;
    float lw[LORA_R];
    #pragma unroll
    for (int t = 0; t < LORA_R; ++t) lw[t] = l2[(size_t)col * LORA_R + t];
    #pragma unroll
    for (int mi = 0; mi < 2; ++mi) {
      #pragma unroll
      for (int r = 0; r < 8; ++r) {
        const int tok = m0 + mi * 16 + r + 8 * mh;
        float val = acc[mi][c][r];
        float ls = 0.f;
        #pragma unroll
        for (int t = 0; t < LORA_R; ++t) ls += xa1[(size_t)tok * LORA_R + t] * lw[t];
        val += LORA_SCALING * ls;
        const int b   = tok >> 11;
        const int ntk = tok & (SEQ - 1);
        const int which = col >> 10;              // 0:q 1:k 2:v
        const int cc = col & (DIM - 1);
        const int h = cc >> 6, d = cc & 63;
        const size_t bh = (size_t)b * NHEADS + h;
        if (which == 0)      q[(bh * SEQ + ntk) * HDIM + d] = (_Float16)(val * ATT_SCALE);
        else if (which == 1) k[(bh * SEQ + ntk) * HDIM + d] = (_Float16)val;
        else                 vT[(bh * HDIM + d) * SEQ + ntk] = (_Float16)val;
      }
    }
  }
}

// ---------------------------------------------------------------------------
// Flash attention: one wave owns a 16-query tile of one (b,h); online softmax
// over 32 keys per iteration (2 S tiles), P bounced through LDS to convert
// the C-fragment layout into an A-fragment for the P@V WMMAs.
// ---------------------------------------------------------------------------
__global__ __launch_bounds__(128) void flash_attn(const _Float16* __restrict__ q,
                                                  const _Float16* __restrict__ k,
                                                  const _Float16* __restrict__ vT,
                                                  _Float16* __restrict__ attn_out) {
  const int bh   = blockIdx.y;                   // b*16 + h
  const int wv   = threadIdx.x >> 5;             // wave in block (0..3)
  const int q0   = (blockIdx.x * 4 + wv) * 16;   // first query row of tile
  const int lane = threadIdx.x & 31;

  const _Float16* qp = q  + (size_t)bh * SEQ * HDIM;
  const _Float16* kp = k  + (size_t)bh * SEQ * HDIM;
  const _Float16* vp = vT + (size_t)bh * HDIM * SEQ;

  __shared__ _Float16 lds_p[4][16 * 32];
  _Float16* pbuf = lds_p[wv];

  const v16h qa0 = load_a_frag(qp + (size_t)q0 * HDIM, HDIM);
  const v16h qa1 = load_a_frag(qp + (size_t)q0 * HDIM + 32, HDIM);

  v8f o[4] = {v8f{}, v8f{}, v8f{}, v8f{}};       // O tile 16x64 (4 n-chunks)
  float mrow[8], lrow[8];
  #pragma unroll
  for (int r = 0; r < 8; ++r) { mrow[r] = -1e30f; lrow[r] = 0.f; }

  const int nloc = lane & 15, mh = lane >> 4;

  for (int j = 0; j < SEQ; j += 32) {
    v16h kb00 = load_bT_frag(kp + (size_t)j * HDIM,             HDIM);
    v16h kb01 = load_bT_frag(kp + (size_t)j * HDIM + 32,        HDIM);
    v16h kb10 = load_bT_frag(kp + (size_t)(j + 16) * HDIM,      HDIM);
    v16h kb11 = load_bT_frag(kp + (size_t)(j + 16) * HDIM + 32, HDIM);
    v8f s0 = v8f{}, s1 = v8f{};
    s0 = wmma_f16(qa0, kb00, s0);  s0 = wmma_f16(qa1, kb01, s0);
    s1 = wmma_f16(qa0, kb10, s1);  s1 = wmma_f16(qa1, kb11, s1);

    float alpha[8];
    #pragma unroll
    for (int r = 0; r < 8; ++r) {
      float mx = fmaxf(s0[r], s1[r]);
      #pragma unroll
      for (int sh = 1; sh < 16; sh <<= 1) mx = fmaxf(mx, __shfl_xor(mx, sh, 32));
      const float mnew = fmaxf(mrow[r], mx);
      alpha[r] = __expf(mrow[r] - mnew);
      mrow[r]  = mnew;
      const float p0 = __expf(s0[r] - mnew);
      const float p1 = __expf(s1[r] - mnew);
      s0[r] = p0;  s1[r] = p1;
      float rs = p0 + p1;
      #pragma unroll
      for (int sh = 1; sh < 16; sh <<= 1) rs += __shfl_xor(rs, sh, 32);
      lrow[r] = lrow[r] * alpha[r] + rs;
    }

    #pragma unroll
    for (int c = 0; c < 4; ++c)
      #pragma unroll
      for (int r = 0; r < 8; ++r) o[c][r] *= alpha[r];

    #pragma unroll
    for (int r = 0; r < 8; ++r) {
      const int m = r + 8 * mh;
      pbuf[m * 32 + nloc]      = (_Float16)s0[r];
      pbuf[m * 32 + 16 + nloc] = (_Float16)s1[r];
    }
    V16H pa;
    {
      const int m = lane & 15, kh = lane >> 4;
      pa.h[0] = *(const v8h*)(pbuf + m * 32 + kh * 8);
      pa.h[1] = *(const v8h*)(pbuf + m * 32 + 16 + kh * 8);
    }

    #pragma unroll
    for (int c = 0; c < 4; ++c) {
      v16h vb = load_bT_frag(vp + (size_t)(c * 16) * SEQ + j, SEQ);
      o[c] = wmma_f16(pa.v, vb, o[c]);
    }
  }

  const int b = bh >> 4, h = bh & 15;
  #pragma unroll
  for (int c = 0; c < 4; ++c)
    #pragma unroll
    for (int r = 0; r < 8; ++r) {
      const int tok = q0 + r + 8 * mh;
      const float val = o[c][r] / lrow[r];
      attn_out[((size_t)b * SEQ + tok) * DIM + h * HDIM + c * 16 + nloc] = (_Float16)val;
    }
}

// ---------------------------------------------------------------------------
// Output projection: out = attn @ proj_w^T + bias + LORA_SCALING * xa2 @ l2^T
// Same cooperative tiling; staging via async global->LDS loads.
// ---------------------------------------------------------------------------
__global__ __launch_bounds__(256) void proj_gemm(const _Float16* __restrict__ ah,
                                                 const _Float16* __restrict__ wh,
                                                 const float* __restrict__ bias,
                                                 const float* __restrict__ xa2,
                                                 const float* __restrict__ l2, // [1024][8]
                                                 float* __restrict__ out) {
  const int NBLK = DIM / 64;                      // 16
  const int mb = blockIdx.x / NBLK;
  const int nb = blockIdx.x % NBLK;
  const int wv = threadIdx.x >> 5;
  const int m0 = mb * 256 + wv * 32;
  const int n0 = nb * 64;

  __shared__ _Float16 bbuf[2][64 * 32];

  stage_w_tile<DIM, 0>(wh, n0, 0, bbuf[0]);
  v8f acc[2][4] = {};
  for (int kk = 0; kk < DIM; kk += 32) {
    const int cur = (kk >> 5) & 1;
    const bool has_next = (kk + 32) < DIM;
    if (has_next) stage_w_tile<DIM, 0>(wh, n0, kk + 32, bbuf[cur ^ 1]);
    stage_wait<0>(has_next);
    __syncthreads();
    const _Float16* ap = ah + (size_t)m0 * DIM + kk;
    __builtin_prefetch(ap + 32, 0, 1);
    v16h a0 = load_a_frag(ap, DIM);
    v16h a1 = load_a_frag(ap + (size_t)16 * DIM, DIM);
    #pragma unroll
    for (int c = 0; c < 4; ++c) {
      v16h b = load_bT_frag(bbuf[cur] + c * 16 * 32, 32);
      acc[0][c] = wmma_f16(a0, b, acc[0][c]);
      acc[1][c] = wmma_f16(a1, b, acc[1][c]);
    }
    __syncthreads();
  }

  const int lane = threadIdx.x & 31;
  const int nloc = lane & 15, mh = lane >> 4;
  #pragma unroll
  for (int c = 0; c < 4; ++c) {
    const int col = n0 + c * 16 + nloc;
    const float bb = bias[col];
    float lw[LORA_R];
    #pragma unroll
    for (int t = 0; t < LORA_R; ++t) lw[t] = l2[(size_t)col * LORA_R + t];
    #pragma unroll
    for (int mi = 0; mi < 2; ++mi) {
      #pragma unroll
      for (int r = 0; r < 8; ++r) {
        const int tok = m0 + mi * 16 + r + 8 * mh;
        float ls = 0.f;
        #pragma unroll
        for (int t = 0; t < LORA_R; ++t) ls += xa2[(size_t)tok * LORA_R + t] * lw[t];
        out[(size_t)tok * DIM + col] = acc[mi][c][r] + bb + LORA_SCALING * ls;
      }
    }
  }
}

// ---------------------------------------------------------------------------
// Host-side orchestration
// ---------------------------------------------------------------------------
extern "C" void kernel_launch(void* const* d_in, const int* in_sizes, int n_in,
                              void* d_out, int out_size, void* d_ws, size_t ws_size,
                              hipStream_t stream) {
  const float* x        = (const float*)d_in[0];   // [2,2048,1024]
  const float* qkv_w    = (const float*)d_in[1];   // [3072,1024]
  const float* proj_w   = (const float*)d_in[2];   // [1024,1024]
  const float* proj_b   = (const float*)d_in[3];   // [1024]
  const float* lw1_l1   = (const float*)d_in[4];   // [8,1024]
  const float* lw1_l2   = (const float*)d_in[5];   // [3072,8]
  const float* lw2_l1   = (const float*)d_in[6];   // [8,1024]
  const float* lw2_l2   = (const float*)d_in[7];   // [1024,8]
  float* out = (float*)d_out;                      // [2,2048,1024]

  char* ws = (char*)d_ws;
  const size_t MB = 1u << 20;
  _Float16* x_h     = (_Float16*)(ws + 0 * MB);    // 8 MiB
  _Float16* qkvw_h  = (_Float16*)(ws + 8 * MB);    // 6 MiB
  _Float16* projw_h = (_Float16*)(ws + 14 * MB);   // 2 MiB
  _Float16* q_h     = (_Float16*)(ws + 16 * MB);   // 8 MiB  [B,H,N,D]
  _Float16* k_h     = (_Float16*)(ws + 24 * MB);   // 8 MiB  [B,H,N,D]
  _Float16* vT_h    = (_Float16*)(ws + 32 * MB);   // 8 MiB  [B,H,D,N]
  _Float16* attn_h  = (_Float16*)(ws + 40 * MB);   // 8 MiB  [B,N,C]
  float*    xa1     = (float*)(ws + 48 * MB);
  float*    xa2     = (float*)(ws + 48 * MB + 256 * 1024);

  cvt_f32_to_f16<<<(NTOK * DIM + 255) / 256, 256, 0, stream>>>(x, x_h, NTOK * DIM);
  cvt_f32_to_f16<<<(QKV_DIM * DIM + 255) / 256, 256, 0, stream>>>(qkv_w, qkvw_h, QKV_DIM * DIM);
  cvt_f32_to_f16<<<(DIM * DIM + 255) / 256, 256, 0, stream>>>(proj_w, projw_h, DIM * DIM);

  lora_down<<<(NTOK * LORA_R + 255) / 256, 256, 0, stream>>>(x_h, lw1_l1, xa1);

  qkv_gemm<<<(NTOK / 256) * (QKV_DIM / 64), 256, 0, stream>>>(x_h, qkvw_h, xa1, lw1_l2,
                                                              q_h, k_h, vT_h);

  {
    dim3 grid(SEQ / 64, 2 * NHEADS);               // (32, 32)
    flash_attn<<<grid, 128, 0, stream>>>(q_h, k_h, vT_h, attn_h);
  }

  lora_down<<<(NTOK * LORA_R + 255) / 256, 256, 0, stream>>>(attn_h, lw2_l1, xa2);

  proj_gemm<<<(NTOK / 256) * (DIM / 64), 256, 0, stream>>>(attn_h, projw_h, proj_b,
                                                           xa2, lw2_l2, out);
}
